// OptimizedSparseAttention_91216515432604
// MI455X (gfx1250) — matrix operand order repaired and checked by reading, not compiled, not run on hardware
//
#include <hip/hip_runtime.h>
#include <math.h>

// ---------------- problem constants (fixed by reference) ----------------
#define B_    2
#define S_    2048
#define D_    1024
#define H_    16
#define DH_   64
#define IH_   4
#define ID_   64
#define TOPK_ 512
#define M_    (B_ * S_)      // 4096 token rows
#define SW_   (S_ / 32)      // 64 mask words per row

// ---------------- CDNA5 WMMA fragment types ----------------
typedef __attribute__((ext_vector_type(16))) _Float16 v16h;
typedef __attribute__((ext_vector_type(8)))  _Float16 v8h;
typedef __attribute__((ext_vector_type(8)))  float    v8f;

// D(16x16,f32) = A(16x32,f16) x B(32x16,f16) + C
__device__ __forceinline__ v8f wmma16(v16h a, v16h b, v8f c) {
  return __builtin_amdgcn_wmma_f32_16x16x32_f16(false, a, false, b, (short)0, c,
                                                false, false);
}

// A-fragment (ISA 7.12.2, 16-bit A 16x32): lane lo holds row M=lo;
// half i<8 -> K = i + 8*hi ; half i>=8 -> K = 16 + (i-8) + 8*hi.
__device__ __forceinline__ v16h frag_a_load(const _Float16* A, int lda, int m0, int k0) {
  const int l = threadIdx.x & 31, lo = l & 15, hi = l >> 4;
  const _Float16* p = A + (size_t)(m0 + lo) * lda + k0 + 8 * hi;
  v8h c0 = *(const v8h*)(p);
  v8h c1 = *(const v8h*)(p + 16);
  v16h r;
#pragma unroll
  for (int i = 0; i < 8; ++i) { r[i] = c0[i]; r[i + 8] = c1[i]; }
  return r;
}

// B-fragment (16-bit B 32x16): lane holds column N = lo;
// half i<8 -> K = i + 8*hi ; half i>=8 -> K = 16 + (i-8) + 8*hi.
__device__ __forceinline__ v16h frag_b_load(const _Float16* Bm, int ldb, int k0, int n0) {
  const int l = threadIdx.x & 31, lo = l & 15, hi = l >> 4;
  const _Float16* p = Bm + (size_t)(k0 + 8 * hi) * ldb + n0 + lo;
  v16h r;
#pragma unroll
  for (int i = 0; i < 8; ++i) { r[i] = p[(size_t)i * ldb]; r[i + 8] = p[(size_t)(16 + i) * ldb]; }
  return r;
}

// ---- CDNA5 async global->LDS copy (ASYNCcnt path, GV addressing mode) ----
// GLOBAL_LOAD_ASYNC_TO_LDS_B128: vdst = per-lane LDS byte offset, vaddr = 64-bit addr.
__device__ __forceinline__ void async_g2l_b128(unsigned lds_off, const _Float16* gaddr) {
  asm volatile("global_load_async_to_lds_b128 %0, %1, off"
               :: "v"(lds_off), "v"(gaddr)
               : "memory");
}
__device__ __forceinline__ void wait_async_le2() {
  asm volatile("s_wait_asynccnt 0x2" ::: "memory");
}
__device__ __forceinline__ void wait_async_0() {
  asm volatile("s_wait_asynccnt 0x0" ::: "memory");
}

// monotone float -> uint key (larger float => larger key)
__device__ __forceinline__ unsigned f2key(float f) {
  unsigned u = __float_as_uint(f);
  return (u & 0x80000000u) ? ~u : (u | 0x80000000u);
}

__device__ __forceinline__ unsigned ballot32(bool p) {
#if __has_builtin(__builtin_amdgcn_ballot_w32)
  return __builtin_amdgcn_ballot_w32(p);
#else
  return (unsigned)__ballot(p);
#endif
}

// ---------------- elementwise casts ----------------
__global__ void k_cast_f16(const float* __restrict__ src, _Float16* __restrict__ dst, int n) {
  int i = blockIdx.x * blockDim.x + threadIdx.x;
  if (i < n) dst[i] = (_Float16)src[i];
}

// ik f32 [B,S,ID] -> ikT f16 [B,ID,S]
__global__ void k_ikT(const float* __restrict__ ikf, _Float16* __restrict__ ikT) {
  int i = blockIdx.x * blockDim.x + threadIdx.x;
  if (i >= B_ * S_ * ID_) return;
  int d = i & (ID_ - 1);
  int s = (i >> 6) & (S_ - 1);
  int b = i >> 17;
  ikT[((size_t)b * ID_ + d) * S_ + s] = (_Float16)ikf[i];
}

// iw = x @ Ww_idx  (N=4, too narrow for a WMMA tile; tiny: 33 MFLOP)
__global__ void k_iw(const float* __restrict__ x, const float* __restrict__ Ww,
                     float* __restrict__ iwf) {
  int i = blockIdx.x * blockDim.x + threadIdx.x;
  if (i >= M_ * IH_) return;
  int h = i & 3, row = i >> 2;
  const float* xr = x + (size_t)row * D_;
  float acc = 0.f;
  for (int k = 0; k < D_; ++k) acc += xr[k] * Ww[k * IH_ + h];
  iwf[i] = acc;
}

// ---------------- f16 WMMA GEMM, compile-time shapes: C[M,N] = A[M,K] @ B[K,N] ----
// block = 128 threads (4 waves); wave w owns N-tile (blockIdx.x*4+w), 2 M-tiles.
// B tiles are double-buffered through LDS with global_load_async_to_lds_b128
// (each lane copies its 32-byte B row chunk), pipelined on ASYNCcnt.
template <int N, int K>
__launch_bounds__(128)
__global__ void k_gemm_t(const _Float16* __restrict__ A, const _Float16* __restrict__ Bm,
                         float* __restrict__ C) {
  __shared__ __align__(16) _Float16 bstage[4][2][32 * 16];  // per-wave dbl buffer, 32x16 f16
  const int w  = threadIdx.x >> 5;
  const int n0 = (blockIdx.x * 4 + w) * 16;
  const int m0 = blockIdx.y * 32;
  const int l = threadIdx.x & 31, lo = l & 15, hi = l >> 4;
  const int krow = lo + 16 * hi;  // B row this lane owns within a 32x16 tile

  const unsigned lds_base =
      (unsigned)(size_t)(&bstage[w][0][0]) + (unsigned)(krow * 16 * sizeof(_Float16));
  const _Float16* gB = Bm + (size_t)krow * N + n0;

  // prologue: stage tile k0=0 into buffer 0
  async_g2l_b128(lds_base, gB);
  async_g2l_b128(lds_base + 16, gB + 8);

  v8f c0 = {};
  v8f c1 = {};
  for (int k0 = 0; k0 < K; k0 += 32) {
    const int buf = (k0 >> 5) & 1;
    if (k0 + 32 < K) {  // issue next B tile into the other buffer, then wait current
      const _Float16* gnext = gB + (size_t)(k0 + 32) * N;
      const unsigned lnext = lds_base + (buf ^ 1) * (unsigned)(32 * 16 * sizeof(_Float16));
      async_g2l_b128(lnext, gnext);
      async_g2l_b128(lnext + 16, gnext + 8);
      __builtin_prefetch((const void*)(A + (size_t)(m0 + lo) * K + k0 + 32), 0, 3);
      wait_async_le2();   // <=2 outstanding => current tile has landed (in-order loads)
    } else {
      wait_async_0();
    }
    // B fragment from LDS (column lo, K = 8*hi + i and 16 + 8*hi + i)
    const _Float16* bp = &bstage[w][buf][(8 * hi) * 16 + lo];
    v16h bf;
#pragma unroll
    for (int i = 0; i < 8; ++i) { bf[i] = bp[i * 16]; bf[i + 8] = bp[(16 + i) * 16]; }

    v16h a0 = frag_a_load(A, K, m0, k0);
    v16h a1 = frag_a_load(A, K, m0 + 16, k0);
    c0 = wmma16(a0, bf, c0);
    c1 = wmma16(a1, bf, c1);
  }
#pragma unroll
  for (int j = 0; j < 8; ++j) {
    C[(size_t)(m0 + j + 8 * hi) * N + n0 + lo]      = c0[j];
    C[(size_t)(m0 + 16 + j + 8 * hi) * N + n0 + lo] = c1[j];
  }
}

// ---------------- RoPE + pack Q/K/V to f16 attention layouts ----------------
// qkv f32 [B,S,3*D] -> Qh [B,H,S,Dh], KtH [B,H,Dh,S] (transposed), Vh [B,H,S,Dh]
__global__ void k_rope_pack(const float* __restrict__ qkv, _Float16* __restrict__ Qh,
                            _Float16* __restrict__ KtH, _Float16* __restrict__ Vh) {
  const int idx = blockIdx.x;
  const int h = idx & (H_ - 1);
  const int s = (idx >> 4) & (S_ - 1);
  const int b = idx >> 15;
  const int tid = threadIdx.x;  // 64
  const float* base = qkv + (size_t)(b * S_ + s) * (3 * D_) + h * DH_;
  const size_t orow = ((size_t)(b * H_ + h) * S_ + s) * DH_;
  Vh[orow + tid] = (_Float16)base[2 * D_ + tid];
  if (tid < 32) {
    // theta_i = 10000^(-2i/Dh), interleaved-pair RoPE (torchtune convention)
    float inv_freq = __expf(-(float)(2 * tid) * (1.0f / 64.0f) * 9.210340371976184f);
    float ang = (float)s * inv_freq;
    float sn, cs;
    __sincosf(ang, &sn, &cs);
    float q0 = base[2 * tid], q1 = base[2 * tid + 1];
    Qh[orow + 2 * tid]     = (_Float16)(q0 * cs - q1 * sn);
    Qh[orow + 2 * tid + 1] = (_Float16)(q1 * cs + q0 * sn);
    float k0 = base[D_ + 2 * tid], k1 = base[D_ + 2 * tid + 1];
    const size_t tb = (size_t)(b * H_ + h) * DH_;
    KtH[(tb + 2 * tid) * S_ + s]     = (_Float16)(k0 * cs - k1 * sn);
    KtH[(tb + 2 * tid + 1) * S_ + s] = (_Float16)(k1 * cs + k0 * sn);
  }
}

// ---------------- fused indexer scores + exact top-k mask ----------------
// grid (S_/16, B_), block 256 (8 waves). Dynamic LDS: scores[16][2048] f32 + hist[8][256].
#define IDX_SMEM (16 * S_ * 4 + 8 * 256 * 4)
__launch_bounds__(256)
__global__ void k_indexer_topk(const _Float16* __restrict__ iqh, const _Float16* __restrict__ ikT,
                               const float* __restrict__ iwf, unsigned* __restrict__ maskw) {
  extern __shared__ char smem_dyn[];
  float* scores  = (float*)smem_dyn;                       // [16][S_]
  unsigned* hist = (unsigned*)(smem_dyn + 16 * S_ * 4);    // [8][256]
  const int b  = blockIdx.y;
  const int t0 = blockIdx.x * 16;
  const int w  = threadIdx.x >> 5;
  const int l  = threadIdx.x & 31, lo = l & 15, hi = l >> 4;

  // ---- phase A: scores[tl][s] = sum_h iw[t,h]*relu(iq[t,h,:]·ik[s,:]) via WMMA ----
  const int mtile = w >> 1, par = w & 1;           // 4 M-tiles of (t,h) rows, 2 waves each
  const _Float16* Abase = iqh + (size_t)(b * S_ + t0) * (IH_ * ID_);
  const _Float16* Bbase = ikT + (size_t)b * ID_ * S_;
  const int tlA = 4 * mtile + 2 * hi, tlB = tlA + 1;
  float wA[4], wB[4];
#pragma unroll
  for (int h2 = 0; h2 < 4; ++h2) {
    wA[h2] = iwf[(size_t)(b * S_ + t0 + tlA) * IH_ + h2];
    wB[h2] = iwf[(size_t)(b * S_ + t0 + tlB) * IH_ + h2];
  }
  v16h a0 = frag_a_load(Abase, ID_, 16 * mtile, 0);
  v16h a1 = frag_a_load(Abase, ID_, 16 * mtile, 32);
  const int ntiles = blockIdx.x + 1;               // causal: s-tiles 0..t0/16
  for (int st = par; st < ntiles; st += 2) {
    const int s0 = st * 16;
    v8f c = {};
    c = wmma16(a0, frag_b_load(Bbase, S_, 0, s0), c);
    c = wmma16(a1, frag_b_load(Bbase, S_, 32, s0), c);
    float sA = 0.f, sB = 0.f;
#pragma unroll
    for (int j = 0; j < 4; ++j) sA += fmaxf(c[j], 0.f) * wA[j];          // h = j
#pragma unroll
    for (int j = 4; j < 8; ++j) sB += fmaxf(c[j], 0.f) * wB[j - 4];
    scores[tlA * S_ + s0 + lo] = sA;
    scores[tlB * S_ + s0 + lo] = sB;
  }
  __syncthreads();

  // ---- phase B: per-row exact 512th-largest via 4x8-bit LDS radix select ----
  for (int rr = 0; rr < 2; ++rr) {
    const int tl = 2 * w + rr;
    const int t  = t0 + tl;
    const int n  = t + 1;
    unsigned thresh = 0;
    if (n > TOPK_) {
      unsigned prefix = 0;
      int want = TOPK_;
      for (int shift = 24; shift >= 0; shift -= 8) {
        for (int i = l; i < 256; i += 32) hist[w * 256 + i] = 0u;
        __builtin_amdgcn_wave_barrier();
        for (int s = l; s <= t; s += 32) {
          unsigned key = f2key(scores[tl * S_ + s]);
          bool ok = (shift == 24) || ((key >> (shift + 8)) == prefix);
          if (ok) atomicAdd(&hist[w * 256 + ((key >> shift) & 255u)], 1u);
        }
        __builtin_amdgcn_wave_barrier();
        int cum = 0;
        unsigned digit = 0;
        for (int d = 255; d >= 0; --d) {          // all lanes compute identically
          int cnt = (int)hist[w * 256 + d];
          if (cum + cnt >= want) { digit = (unsigned)d; break; }
          cum += cnt;
        }
        want -= cum;
        prefix = (prefix << 8) | digit;
        __builtin_amdgcn_wave_barrier();
      }
      thresh = prefix;
    }
    // emit causal top-k bitmask (wave32 ballot -> one 32-bit word per 32 columns)
    for (int wi = 0; wi < SW_; ++wi) {
      const int s = wi * 32 + l;
      bool sel = (s <= t) && ((n <= TOPK_) || (f2key(scores[tl * S_ + s]) >= thresh));
      unsigned word = ballot32(sel);
      if (l == 0) maskw[((size_t)b * S_ + t) * SW_ + wi] = word;
    }
  }
}

// ---------------- sparse flash attention (WMMA, mask-tile skipping) ----------------
// grid (B_*H_, S_/64), block 128 (4 waves); wave owns 16 t-rows.
__launch_bounds__(128)
__global__ void k_flash_attn(const _Float16* __restrict__ Qh, const _Float16* __restrict__ KtH,
                             const _Float16* __restrict__ Vh, const unsigned* __restrict__ maskw,
                             _Float16* __restrict__ attnh) {
  __shared__ __align__(16) _Float16 pbuf[4][16 * 32];
  const int bh = blockIdx.x;
  const int b = bh >> 4, h = bh & 15;
  const int w = threadIdx.x >> 5;
  const int t0 = (blockIdx.y * 4 + w) * 16;
  const int l = threadIdx.x & 31, lo = l & 15, hi = l >> 4;
  const float scale = 0.125f;  // 1/sqrt(64)

  const _Float16* Qbase = Qh + ((size_t)bh * S_ + t0) * DH_;
  const _Float16* Ktb   = KtH + (size_t)bh * DH_ * S_;
  const _Float16* Vb    = Vh + (size_t)bh * S_ * DH_;
  const unsigned* mrow  = maskw + ((size_t)b * S_ + t0) * SW_;

  v16h aq0 = frag_a_load(Qbase, DH_, 0, 0);
  v16h aq1 = frag_a_load(Qbase, DH_, 0, 32);
  float m[8], lsum[8];
  v8f acc[4];
  v8f vzero = {};
#pragma unroll
  for (int j = 0; j < 8; ++j) { m[j] = -3.0e38f; lsum[j] = 0.f; }
#pragma unroll
  for (int nt = 0; nt < 4; ++nt) acc[nt] = vzero;

  for (int s0 = 0; s0 <= t0 + 15; s0 += 32) {
    // mask words for the 16 rows x 32 columns strip
    unsigned mw = 0;
    if (l < 16) mw = mrow[(size_t)l * SW_ + (s0 >> 5)];
    unsigned anyw = mw;
#pragma unroll
    for (int off = 16; off >= 1; off >>= 1) anyw |= __shfl_xor(anyw, off, 32);
    if (anyw == 0) continue;  // top-k sparsity: skip fully-masked strip

    // logits for two 16-column subtiles
    v8f c0 = {}, c1 = {};
    c0 = wmma16(aq0, frag_b_load(Ktb, S_, 0, s0), c0);
    c0 = wmma16(aq1, frag_b_load(Ktb, S_, 32, s0), c0);
    c1 = wmma16(aq0, frag_b_load(Ktb, S_, 0, s0 + 16), c1);
    c1 = wmma16(aq1, frag_b_load(Ktb, S_, 32, s0 + 16), c1);

#pragma unroll
    for (int j = 0; j < 8; ++j) {
      unsigned wj = (unsigned)__shfl((int)mw, j + 8 * hi, 32);
      bool sel0 = (wj >> lo) & 1u;
      bool sel1 = (wj >> (16 + lo)) & 1u;
      float v0 = sel0 ? c0[j] * scale : -3.0e38f;
      float v1 = sel1 ? c1[j] * scale : -3.0e38f;
      float mt = fmaxf(v0, v1);
#pragma unroll
      for (int off = 1; off < 16; off <<= 1) mt = fmaxf(mt, __shfl_xor(mt, off, 32));
      float mnew = fmaxf(m[j], mt);
      float alpha, p0, p1;
      if (mnew <= -1.0e37f) { alpha = 1.f; p0 = 0.f; p1 = 0.f; }
      else {
        alpha = __expf(m[j] - mnew);
        p0 = sel0 ? __expf(v0 - mnew) : 0.f;
        p1 = sel1 ? __expf(v1 - mnew) : 0.f;
      }
      float rs = p0 + p1;
#pragma unroll
      for (int off = 1; off < 16; off <<= 1) rs += __shfl_xor(rs, off, 32);
      lsum[j] = lsum[j] * alpha + rs;
      m[j] = mnew;
#pragma unroll
      for (int nt = 0; nt < 4; ++nt) acc[nt][j] *= alpha;
      pbuf[w][(j + 8 * hi) * 32 + lo]      = (_Float16)p0;
      pbuf[w][(j + 8 * hi) * 32 + 16 + lo] = (_Float16)p1;
    }
    __builtin_amdgcn_wave_barrier();

    // re-fragment P from LDS into A-layout (row t=lo, K=s_local), then P @ V
    const _Float16* pp = &pbuf[w][lo * 32 + 8 * hi];
    v8h p0v = *(const v8h*)(pp);
    v8h p1v = *(const v8h*)(pp + 16);
    v16h ap;
#pragma unroll
    for (int i = 0; i < 8; ++i) { ap[i] = p0v[i]; ap[i + 8] = p1v[i]; }
#pragma unroll
    for (int nt = 0; nt < 4; ++nt)
      acc[nt] = wmma16(ap, frag_b_load(Vb, DH_, s0, nt * 16), acc[nt]);
  }

  // epilogue: normalize and scatter f16 into attn [B,S,D] (col = h*64 + d)
#pragma unroll
  for (int j = 0; j < 8; ++j) {
    float inv = lsum[j] > 0.f ? 1.0f / lsum[j] : 0.f;
    const int t = t0 + j + 8 * hi;
    const size_t orow = ((size_t)b * S_ + t) * D_ + h * DH_;
#pragma unroll
    for (int nt = 0; nt < 4; ++nt)
      attnh[orow + nt * 16 + lo] = (_Float16)(acc[nt][j] * inv);
  }
}

// ---------------- host orchestration ----------------
extern "C" void kernel_launch(void* const* d_in, const int* in_sizes, int n_in,
                              void* d_out, int out_size, void* d_ws, size_t ws_size,
                              hipStream_t stream) {
  (void)in_sizes; (void)n_in; (void)out_size; (void)ws_size;
  const float* x    = (const float*)d_in[0];
  const float* Wqkv = (const float*)d_in[1];
  const float* Wo   = (const float*)d_in[2];
  const float* Wqi  = (const float*)d_in[3];
  const float* Wki  = (const float*)d_in[4];
  const float* Wwi  = (const float*)d_in[5];
  float* out = (float*)d_out;   // [B,S,D] fp32

  char* ws = (char*)d_ws;
  size_t off = 0;
  auto alloc = [&](size_t bytes) -> char* {
    char* p = ws + off;
    off += (bytes + 255) & ~(size_t)255;
    return p;
  };
  _Float16* xh    = (_Float16*)alloc((size_t)M_ * D_ * 2);
  _Float16* Wqkvh = (_Float16*)alloc((size_t)D_ * 3 * D_ * 2);
  _Float16* Woh   = (_Float16*)alloc((size_t)D_ * D_ * 2);
  _Float16* Wqih  = (_Float16*)alloc((size_t)D_ * IH_ * ID_ * 2);
  _Float16* Wkih  = (_Float16*)alloc((size_t)D_ * ID_ * 2);
  float*    qkvf  = (float*)alloc((size_t)M_ * 3 * D_ * 4);
  _Float16* Qh    = (_Float16*)alloc((size_t)M_ * D_ * 2);
  _Float16* Kth   = (_Float16*)alloc((size_t)M_ * D_ * 2);
  _Float16* Vh    = (_Float16*)alloc((size_t)M_ * D_ * 2);
  float*    iqf   = (float*)alloc((size_t)M_ * IH_ * ID_ * 4);
  _Float16* iqh   = (_Float16*)alloc((size_t)M_ * IH_ * ID_ * 2);
  float*    ikf   = (float*)alloc((size_t)M_ * ID_ * 4);
  _Float16* ikth  = (_Float16*)alloc((size_t)B_ * ID_ * S_ * 2);
  float*    iwf   = (float*)alloc((size_t)M_ * IH_ * 4);
  unsigned* maskw = (unsigned*)alloc((size_t)B_ * S_ * SW_ * 4);
  _Float16* attnh = (_Float16*)alloc((size_t)M_ * D_ * 2);

  auto cast = [&](const float* s, _Float16* d, int n) {
    k_cast_f16<<<dim3((n + 255) / 256), dim3(256), 0, stream>>>(s, d, n);
  };
  cast(x, xh, M_ * D_);
  cast(Wqkv, Wqkvh, D_ * 3 * D_);
  cast(Wo, Woh, D_ * D_);
  cast(Wqi, Wqih, D_ * IH_ * ID_);
  cast(Wki, Wkih, D_ * ID_);

  // QKV projection (WMMA + async B-tile staging)
  k_gemm_t<3 * D_, D_><<<dim3(3 * D_ / 64, M_ / 32), dim3(128), 0, stream>>>(xh, Wqkvh, qkvf);
  // RoPE + pack Q/K^T/V
  k_rope_pack<<<dim3(B_ * S_ * H_), dim3(64), 0, stream>>>(qkvf, Qh, Kth, Vh);

  // indexer projections (WMMA) + small iw
  k_gemm_t<IH_ * ID_, D_><<<dim3(IH_ * ID_ / 64, M_ / 32), dim3(128), 0, stream>>>(xh, Wqih, iqf);
  cast(iqf, iqh, M_ * IH_ * ID_);
  k_gemm_t<ID_, D_><<<dim3(ID_ / 64, M_ / 32), dim3(128), 0, stream>>>(xh, Wkih, ikf);
  k_ikT<<<dim3((B_ * S_ * ID_ + 255) / 256), dim3(256), 0, stream>>>(ikf, ikth);
  k_iw<<<dim3((M_ * IH_ + 255) / 256), dim3(256), 0, stream>>>(x, Wwi, iwf);

  // fused indexer scores (WMMA) + exact top-512 mask (LDS radix select)
  (void)hipFuncSetAttribute((const void*)k_indexer_topk,
                            hipFuncAttributeMaxDynamicSharedMemorySize, IDX_SMEM);
  k_indexer_topk<<<dim3(S_ / 16, B_), dim3(256), IDX_SMEM, stream>>>(iqh, ikth, iwf, maskw);

  // sparse flash attention (WMMA, tile skipping via mask)
  k_flash_attn<<<dim3(B_ * H_, S_ / 64), dim3(128), 0, stream>>>(Qh, Kth, Vh, maskw, attnh);

  // output projection (WMMA) -> fp32 result
  k_gemm_t<D_, D_><<<dim3(D_ / 64, M_ / 32), dim3(128), 0, stream>>>(attnh, Woh, out);
}
